// get_model_35725537968322
// MI455X (gfx1250) — compile-verified, matrix operand order checked by reading, not run on hardware
//
#include <hip/hip_runtime.h>
#include <hip/hip_bf16.h>

typedef __attribute__((ext_vector_type(16))) _Float16 v16h;
typedef __attribute__((ext_vector_type(8)))  float    v8f;

#define BN_INV 0.9999950000374997f   // 1/sqrt(1+1e-5)

// f16 weights, pre-padded: cin padded to multiple of 32 with zeros
struct LayerH { const _Float16* w; const float* b; const float* g; const float* be; int cinPad; int cout; };

// ---------------- WMMA fragment helpers (CDNA5 wave32 layouts, ISA 7.12.2) -----------

// A-matrix 16x32 f16 from LDS: lanes 0-15 -> rows, K 0..15 ; lanes 16-31 -> rows, K 16..31
__device__ __forceinline__ v16h frag_a_lds(const _Float16* X, int pitch, int m0, int k0, int lane){
  int row = m0 + (lane & 15);
  int kb  = k0 + ((lane >> 4) << 4);
  v16h a;
#pragma unroll
  for (int i = 0; i < 16; ++i) a[i] = X[row * pitch + kb + i];
  return a;
}

// B-matrix 32x16 f16 directly from padded-f16 global weights: one aligned 32B vector load
// (stride and k0 are multiples of 32/16 halves -> 32B alignment guaranteed)
__device__ __forceinline__ v16h frag_b_h(const _Float16* __restrict__ wh, int stride, int woff,
                                         int n0, int k0, int lane){
  int col = n0 + (lane & 15);
  int kb  = k0 + ((lane >> 4) << 4);
  return *(const v16h*)(wh + (size_t)col * stride + woff + kb);
}

#define WMMA16(a, b, c) __builtin_amdgcn_wmma_f32_16x16x32_f16(false, a, false, b, (short)0, c, false, false)

// ---------------- weight pre-conversion f32 -> padded f16 ----------------------------
__global__ void w2h_kernel(const float* __restrict__ w, _Float16* __restrict__ wh,
                           int cout, int cin, int cinPad){
  int t = blockIdx.x * blockDim.x + threadIdx.x;
  if (t >= cout * cinPad) return;
  int c = t % cinPad, o = t / cinPad;
  wh[t] = (c < cin) ? (_Float16)w[(size_t)o * cin + c] : (_Float16)0.f;
}

// ---------------- generic tiled WMMA GEMM:  Y = epilogue(X[M,K] @ W[N,K]^T) ----------
// Block tile 128(M) x 64(N), 8 wave32s, wave tile 32x32 (4 WMMAs / K-chunk).
// Requires: N multiple of 64 (true for all call sites). mode 0: raw, 1: relu(bn(x+bias)).
__global__ __launch_bounds__(256)
void gemm_f16w_kernel(const float* __restrict__ X, int ldx,
                      const _Float16* __restrict__ Wh, int wstride, int woff,
                      const float* __restrict__ bias, const float* __restrict__ g,
                      const float* __restrict__ be,
                      float* __restrict__ Y, int ldy, int ycoloff,
                      int M, int K, int N, int mode){
  __shared__ _Float16 As[128 * 40];
  const int tid = threadIdx.x, lane = tid & 31, wv = tid >> 5;
  const int m0 = blockIdx.x * 128, n0 = blockIdx.y * 64;
  const int wm = (wv & 3) * 32, wn = (wv >> 2) * 32;
  v8f acc00 = {}, acc01 = {}, acc10 = {}, acc11 = {};
  const int kc = (K + 31) >> 5;
  const int r = tid >> 1, cb = (tid & 1) * 16;   // each thread stages 16 contiguous elems
  for (int kch = 0; kch < kc; ++kch){
    const int k0 = kch * 32;
    if (m0 + 128 <= M && k0 + 32 <= K){
      const float* src = X + (size_t)(m0 + r) * ldx + k0 + cb;
      if (kch + 1 < kc) __builtin_prefetch(src + 32, 0, 1);   // global_prefetch next chunk
#pragma unroll
      for (int i = 0; i < 16; ++i) As[r * 40 + cb + i] = (_Float16)src[i];
    } else {
#pragma unroll
      for (int i = 0; i < 16; ++i){
        int gr = m0 + r, gk = k0 + cb + i;
        As[r * 40 + cb + i] = (gr < M && gk < K) ? (_Float16)X[(size_t)gr * ldx + gk]
                                                 : (_Float16)0.f;
      }
    }
    __syncthreads();
    v16h a0 = frag_a_lds(As, 40, wm, 0, lane);
    v16h a1 = frag_a_lds(As, 40, wm + 16, 0, lane);
    v16h b0 = frag_b_h(Wh, wstride, woff, n0 + wn, k0, lane);
    v16h b1 = frag_b_h(Wh, wstride, woff, n0 + wn + 16, k0, lane);
    acc00 = WMMA16(a0, b0, acc00);
    acc01 = WMMA16(a0, b1, acc01);
    acc10 = WMMA16(a1, b0, acc10);
    acc11 = WMMA16(a1, b1, acc11);
    __syncthreads();
  }
  const int rlo = ((lane >> 4) << 3), cl = (lane & 15);
#pragma unroll
  for (int msub = 0; msub < 2; ++msub){
#pragma unroll
    for (int nsub = 0; nsub < 2; ++nsub){
      v8f acc = (msub == 0) ? (nsub == 0 ? acc00 : acc01) : (nsub == 0 ? acc10 : acc11);
      int col = n0 + wn + nsub * 16 + cl;        // col < N guaranteed (N % 64 == 0)
#pragma unroll
      for (int j = 0; j < 8; ++j){
        int row = m0 + wm + msub * 16 + j + rlo;
        if (row < M){
          float v = acc[j];
          if (mode >= 1){
            v += bias[col];
            v = v * (g[col] * BN_INV) + be[col];
            if (v < 0.f) v = 0.f;
          }
          Y[(size_t)row * ldy + ycoloff + col] = v;
        }
      }
    }
  }
}

// ---------------- fused 3-layer MLP + max-pool over group (SA branches) --------------
__device__ void run_layer(const _Float16* Xin, int pin, _Float16* Xout, int pout,
                          LayerH L, int Mpad){
  const int tid = threadIdx.x, lane = tid & 31, wv = tid >> 5;
  const int mtiles = Mpad >> 4, ntiles = L.cout >> 4;
  const int tiles = mtiles * ntiles;
  const int kc = L.cinPad >> 5;
  for (int t = wv; t < tiles; t += 8){
    int mt = t % mtiles, nt = t / mtiles;
    v8f acc = {};
    for (int k = 0; k < kc; ++k){
      v16h a = frag_a_lds(Xin, pin, mt * 16, k * 32, lane);
      v16h b = frag_b_h(L.w, L.cinPad, 0, nt * 16, k * 32, lane);
      acc = WMMA16(a, b, acc);
    }
#pragma unroll
    for (int j = 0; j < 8; ++j){
      int row = mt * 16 + j + ((lane >> 4) << 3);
      int col = nt * 16 + (lane & 15);
      float v = acc[j] + L.b[col];
      v = v * (L.g[col] * BN_INV) + L.be[col];
      if (v < 0.f) v = 0.f;
      Xout[row * pout + col] = (_Float16)v;
    }
  }
  __syncthreads();
}

__global__ __launch_bounds__(256)
void branch_mlp_kernel(const float* __restrict__ X0, int K,
                       LayerH L1, LayerH L2, LayerH L3,
                       float* __restrict__ out, int ostride, int ocol){
  __shared__ _Float16 bufA[128 * 96];    // 24KB, inputs up to 96 ch
  __shared__ _Float16 bufB[128 * 128];   // 32KB, outputs up to 128 ch
  const int tid = threadIdx.x;
  const int Mpad = (K <= 16) ? 16 : 128;
  const float* x0 = X0 + (size_t)blockIdx.x * K * 6;
  for (int e = tid; e < Mpad * 32; e += 256){        // zero the padded input tile
    int r = e >> 5, c = e & 31;
    bufA[r * 96 + c] = (_Float16)0.f;
  }
  __syncthreads();
  for (int e = tid; e < K * 6; e += 256){            // unguarded contiguous fill
    int r = e / 6, c = e % 6;
    bufA[r * 96 + c] = (_Float16)x0[e];
  }
  __syncthreads();
  run_layer(bufA, 96, bufB, 128, L1, Mpad);
  run_layer(bufB, 128, bufA, 96, L2, Mpad);
  run_layer(bufA, 96, bufB, 128, L3, Mpad);
  for (int c = tid; c < L3.cout; c += 256){
    float m = -1e30f;
    for (int r = 0; r < K; ++r){ float v = (float)bufB[r * 128 + c]; if (v > m) m = v; }
    out[(size_t)blockIdx.x * ostride + ocol + c] = m;
  }
}

// ---------------- fused pscn 'uv' layer: build relu(bn(P+Q)) rows, GEMM, max over s ---
__global__ __launch_bounds__(256)
void pscn_uv_kernel(const float* __restrict__ Pu, const float* __restrict__ Qu,
                    const float* __restrict__ Pv, const float* __restrict__ Qv,
                    const float* gub, const float* gug, const float* gube,
                    const float* gvb, const float* gvg, const float* gvbe,
                    LayerH uv, float* __restrict__ hmax){
  __shared__ _Float16 Xt[64 * 416];      // 52KB: 64 samples x 390 (pad 416) f16
  __shared__ float pmax[4 * 192];
  const int tid = threadIdx.x, lane = tid & 31, wv = tid >> 5;
  const int bn = blockIdx.x;             // b*256 + n
  const int b = bn >> 8;
  const float* pu = Pu + (size_t)bn * 6;
  const float* pv = Pv + (size_t)bn * 384;
  for (int e = tid; e < 64 * 416; e += 256){
    int s = e / 416, c = e % 416;
    float v = 0.f;
    if (c < 6){
      float x = pu[c] + Qu[(b * 64 + s) * 6 + c] + gub[c];
      x = x * (gug[c] * BN_INV) + gube[c];
      v = x > 0.f ? x : 0.f;
    } else if (c < 390){
      int cc = c - 6;
      float x = pv[cc] + Qv[((size_t)b * 64 + s) * 384 + cc] + gvb[cc];
      x = x * (gvg[cc] * BN_INV) + gvbe[cc];
      v = x > 0.f ? x : 0.f;
    }
    Xt[e] = (_Float16)v;
  }
  for (int e = tid; e < 4 * 192; e += 256) pmax[e] = -1e30f;
  __syncthreads();
  for (int t = wv; t < 48; t += 8){      // 4 m-tiles x 12 n-tiles
    int mt = t & 3, nt = t >> 2;
    v8f acc = {};
    for (int k = 0; k < 13; ++k){        // 416 = 13*32, matches weight pad -> unguarded
      v16h a = frag_a_lds(Xt, 416, mt * 16, k * 32, lane);
      v16h bf = frag_b_h(uv.w, 416, 0, nt * 16, k * 32, lane);
      acc = WMMA16(a, bf, acc);
    }
    int col = nt * 16 + (lane & 15);
    float m = -1e30f;
#pragma unroll
    for (int j = 0; j < 8; ++j){
      float v = acc[j] + uv.b[col];
      v = v * (uv.g[col] * BN_INV) + uv.be[col];
      if (v < 0.f) v = 0.f;
      if (v > m) m = v;
    }
    float o = __shfl_xor(m, 16, 32);     // merge the two row-halves (wave32)
    if (o > m) m = o;
    if (lane < 16) pmax[mt * 192 + col] = m;
  }
  __syncthreads();
  for (int c = tid; c < 192; c += 256){
    float m = pmax[c];
    for (int mt = 1; mt < 4; ++mt){ float v = pmax[mt * 192 + c]; if (v > m) m = v; }
    hmax[(size_t)bn * 192 + c] = m;
  }
}

// ---------------- scalar / irregular kernels ----------------------------------------

__global__ void split_transpose_kernel(const float* __restrict__ xyz,
                                       float* __restrict__ coords, float* __restrict__ norm){
  int t = blockIdx.x * blockDim.x + threadIdx.x;
  if (t >= 8 * 2048) return;
  int b = t >> 11, n = t & 2047;
  for (int k = 0; k < 3; ++k){
    coords[(size_t)t * 3 + k] = xyz[((size_t)b * 6 + k) * 2048 + n];
    norm  [(size_t)t * 3 + k] = xyz[((size_t)b * 6 + 3 + k) * 2048 + n];
  }
}

__global__ __launch_bounds__(256)
void fps_kernel(const float* __restrict__ coords, float* __restrict__ dist,
                int* __restrict__ fidx, float* __restrict__ new_xyz){
  __shared__ float sred[256]; __shared__ int sidx[256];
  __shared__ int cur; __shared__ float cen[3];
  const int b = blockIdx.x, tid = threadIdx.x;
  const float* cb = coords + (size_t)b * 2048 * 3;
  float* db = dist + (size_t)b * 2048;
  for (int i = tid; i < 2048; i += 256) db[i] = 1e10f;
  if (tid == 0) cur = 0;
  __syncthreads();
  for (int it = 0; it < 256; ++it){
    if (tid == 0){
      int c = cur;
      fidx[b * 256 + it] = c;
      cen[0] = cb[c * 3]; cen[1] = cb[c * 3 + 1]; cen[2] = cb[c * 3 + 2];
      new_xyz[((size_t)b * 256 + it) * 3 + 0] = cen[0];
      new_xyz[((size_t)b * 256 + it) * 3 + 1] = cen[1];
      new_xyz[((size_t)b * 256 + it) * 3 + 2] = cen[2];
    }
    __syncthreads();
    float best = -1.f; int bi = 0;
    for (int i = tid; i < 2048; i += 256){
      float dx = cb[i * 3] - cen[0], dy = cb[i * 3 + 1] - cen[1], dz = cb[i * 3 + 2] - cen[2];
      float d = dx * dx + dy * dy + dz * dz;
      float nd = db[i]; if (d < nd) nd = d; db[i] = nd;
      if (nd > best){ best = nd; bi = i; }
    }
    sred[tid] = best; sidx[tid] = bi;
    __syncthreads();
    for (int s = 128; s > 0; s >>= 1){
      if (tid < s && sred[tid + s] > sred[tid]){ sred[tid] = sred[tid + s]; sidx[tid] = sidx[tid + s]; }
      __syncthreads();
    }
    if (tid == 0) cur = sidx[0];
    __syncthreads();
  }
}

// first-K-in-index-order ball query with gather of [norm ; xyz-center]
__global__ void ballq_kernel(const float* __restrict__ coords, const float* __restrict__ norm,
                             const float* __restrict__ new_xyz, float* __restrict__ X0,
                             int K, float r2){
  int t = blockIdx.x * blockDim.x + threadIdx.x;
  if (t >= 8 * 256) return;
  int b = t >> 8;
  const float* cb = coords + (size_t)b * 2048 * 3;
  const float* nb = norm + (size_t)b * 2048 * 3;
  float qx = new_xyz[t * 3], qy = new_xyz[t * 3 + 1], qz = new_xyz[t * 3 + 2];
  float* row = X0 + (size_t)t * K * 6;
  int cnt = 0, first = -1;
  for (int n = 0; n < 2048 && cnt < K; ++n){
    float dx = cb[n * 3] - qx, dy = cb[n * 3 + 1] - qy, dz = cb[n * 3 + 2] - qz;
    float d = dx * dx + dy * dy + dz * dz;
    if (d <= r2){
      if (first < 0) first = n;
      float* r = row + cnt * 6;
      r[0] = nb[n * 3]; r[1] = nb[n * 3 + 1]; r[2] = nb[n * 3 + 2];
      r[3] = dx; r[4] = dy; r[5] = dz;
      ++cnt;
    }
  }
  int fn = (first < 0) ? 0 : first;
  float fdx = cb[fn * 3] - qx, fdy = cb[fn * 3 + 1] - qy, fdz = cb[fn * 3 + 2] - qz;
  for (; cnt < K; ++cnt){
    float* r = row + cnt * 6;
    r[0] = nb[fn * 3]; r[1] = nb[fn * 3 + 1]; r[2] = nb[fn * 3 + 2];
    r[3] = fdx; r[4] = fdy; r[5] = fdz;
  }
}

__device__ __forceinline__ unsigned int part1by2(unsigned int x){
  x &= 1023u;
  x = (x | (x << 16)) & 50331903u;
  x = (x | (x << 8))  & 50393103u;
  x = (x | (x << 4))  & 51130563u;
  x = (x | (x << 2))  & 153391689u;
  return x;
}

__global__ __launch_bounds__(256)
void zorder_kernel(const float* __restrict__ new_xyz, int* __restrict__ ides){
  __shared__ unsigned long long key[256];
  __shared__ float mn[3], mx[3];
  const int b = blockIdx.x, t = threadIdx.x;
  const float* u = new_xyz + (size_t)b * 256 * 3;
  if (t == 0){
    for (int k = 0; k < 3; ++k){ mn[k] = 1e30f; mx[k] = -1e30f; }
    for (int i = 0; i < 256; ++i)
      for (int k = 0; k < 3; ++k){
        float v = u[i * 3 + k]; if (v < mn[k]) mn[k] = v; if (v > mx[k]) mx[k] = v;
      }
  }
  __syncthreads();
  unsigned int q[3];
  for (int k = 0; k < 3; ++k){
    float v = (u[t * 3 + k] - mn[k]) / (mx[k] - mn[k] + 1e-9f) * 1023.f;
    int qi = (int)v; if (qi < 0) qi = 0; if (qi > 1023) qi = 1023;
    q[k] = (unsigned int)qi;
  }
  unsigned int code = part1by2(q[0]) | (part1by2(q[1]) << 1) | (part1by2(q[2]) << 2);
  key[t] = ((unsigned long long)code << 32) | (unsigned long long)t;  // stable sort via idx in LSBs
  __syncthreads();
  for (int k = 2; k <= 256; k <<= 1){
    for (int j = k >> 1; j > 0; j >>= 1){
      int ixj = t ^ j;
      if (ixj > t){
        bool up = ((t & k) == 0);
        unsigned long long a = key[t], c = key[ixj];
        if (up ? (a > c) : (a < c)){ key[t] = c; key[ixj] = a; }
      }
      __syncthreads();
    }
  }
  if (t < 64) ides[b * 64 + t] = (int)(key[t * 4] & 0xffffffffull);   // stride = 256/64
}

__global__ void gather_sample_kernel(const float* __restrict__ new_xyz, const float* __restrict__ v,
                                     const int* __restrict__ ides, const float* __restrict__ wgu,
                                     float* __restrict__ su, float* __restrict__ sv,
                                     float* __restrict__ Qu){
  int t = blockIdx.x * blockDim.x + threadIdx.x;
  if (t >= 8 * 64) return;
  int b = t >> 6;
  int idx = ides[t];
  const float* p = new_xyz + ((size_t)b * 256 + idx) * 3;
  su[t * 3 + 0] = p[0]; su[t * 3 + 1] = p[1]; su[t * 3 + 2] = p[2];
  const float* vv = v + ((size_t)b * 256 + idx) * 192;
  for (int c = 0; c < 192; ++c) sv[(size_t)t * 192 + c] = vv[c];
  for (int o = 0; o < 6; ++o){
    float s = 0.f;
    for (int k = 0; k < 3; ++k) s += p[k] * wgu[o * 6 + 3 + k];   // right half of gu weight
    Qu[t * 6 + o] = s;
  }
}

__global__ void pu_kernel(const float* __restrict__ new_xyz, const float* __restrict__ wgu,
                          float* __restrict__ Pu){
  int t = blockIdx.x * blockDim.x + threadIdx.x;
  if (t >= 8 * 256) return;
  const float* p = new_xyz + (size_t)t * 3;
  for (int o = 0; o < 6; ++o){
    float s = 0.f;
    for (int k = 0; k < 3; ++k) s += p[k] * wgu[o * 6 + k];       // left half of gu weight
    Pu[t * 6 + o] = s;
  }
}

__global__ void l1_concat_kernel(const float* __restrict__ v, const float* __restrict__ hf,
                                 const float* __restrict__ new_xyz, float* __restrict__ l1){
  int t = blockIdx.x * blockDim.x + threadIdx.x;
  if (t >= 8 * 256 * 195) return;
  int c = t % 195, bn = t / 195;
  float val = (c < 192) ? v[(size_t)bn * 192 + c] + hf[(size_t)bn * 192 + c]
                        : new_xyz[(size_t)bn * 3 + (c - 192)];
  l1[t] = val;
}

__global__ void meanmax_kernel(const float* __restrict__ l1, float* __restrict__ amean,
                               float* __restrict__ amax){
  int b = blockIdx.x, tid = threadIdx.x;
  for (int c = tid; c < 195; c += 256){
    float s = 0.f, m = -1e30f;
    for (int n = 0; n < 256; ++n){
      float v = l1[((size_t)b * 256 + n) * 195 + c];
      s += v; if (v > m) m = v;
    }
    amean[b * 195 + c] = s * (1.f / 256.f);
    amax[b * 195 + c] = m;
  }
}

__global__ __launch_bounds__(256)
void att_kernel(const float* __restrict__ amean, const float* __restrict__ amax,
                const float* __restrict__ w1a, const float* __restrict__ w2a,
                const float* __restrict__ w1b, const float* __restrict__ w2b,
                float* __restrict__ out, int cout){
  __shared__ float h1[97], h2[97];
  int b = blockIdx.x, tid = threadIdx.x;
  const float* me = amean + b * 195;
  const float* mx = amax + b * 195;
  for (int h = tid; h < 97; h += 256){
    float s1 = 0.f, s2 = 0.f;
    for (int c = 0; c < 195; ++c){ s1 += me[c] * w1a[h * 195 + c]; s2 += mx[c] * w1b[h * 195 + c]; }
    h1[h] = s1 > 0.f ? s1 : 0.f;
    h2[h] = s2 > 0.f ? s2 : 0.f;
  }
  __syncthreads();
  for (int o = tid; o < cout; o += 256){
    float s = 0.f;
    for (int h = 0; h < 97; ++h) s += h1[h] * w2a[o * 97 + h] + h2[h] * w2b[o * 97 + h];
    out[b * cout + o] = 1.f / (1.f + __expf(-s));
  }
}

__global__ void scale_concat_kernel(const float* __restrict__ l1, const float* __restrict__ ch,
                                    const float* __restrict__ sp, const float* __restrict__ new_xyz,
                                    float* __restrict__ h198){
  int t = blockIdx.x * blockDim.x + threadIdx.x;
  if (t >= 8 * 256 * 198) return;
  int c = t % 198, bn = t / 198, b = bn >> 8, n = bn & 255;
  float v;
  if (c < 3) v = new_xyz[(size_t)bn * 3 + c];
  else { int cc = c - 3; v = l1[(size_t)bn * 195 + cc] * ch[b * 195 + cc] * sp[b * 256 + n]; }
  h198[t] = v;
}

__global__ void maxn_kernel(const float* __restrict__ t3, float* __restrict__ l3,
                            float* __restrict__ out_l3){
  int t = blockIdx.x * blockDim.x + threadIdx.x;
  if (t >= 8 * 1024) return;
  int b = t >> 10, c = t & 1023;
  float m = -1e30f;
  for (int n = 0; n < 256; ++n){
    float v = t3[((size_t)b * 256 + n) * 1024 + c];
    if (v > m) m = v;
  }
  l3[t] = m;
  out_l3[t] = m;
}

__global__ void fc_kernel(const float* __restrict__ X, const float* __restrict__ W,
                          const float* __restrict__ bias, const float* __restrict__ g,
                          const float* __restrict__ be, float* __restrict__ Y,
                          int cin, int cout, int mode){
  int b = blockIdx.x;
  int o = blockIdx.y * blockDim.x + threadIdx.x;
  if (o >= cout) return;
  const float* x = X + (size_t)b * cin;
  float s = bias[o];
  for (int c = 0; c < cin; ++c) s += x[c] * W[(size_t)o * cin + c];
  if (mode == 1){ s = s * (g[o] * BN_INV) + be[o]; if (s < 0.f) s = 0.f; }
  Y[(size_t)b * cout + o] = s;
}

__global__ void logsoftmax_kernel(const float* __restrict__ x, float* __restrict__ out){
  __shared__ float mx, ls;
  int b = blockIdx.x, tid = threadIdx.x;
  if (tid == 0){
    float m = -1e30f;
    for (int i = 0; i < 40; ++i){ float v = x[b * 40 + i]; if (v > m) m = v; }
    float s = 0.f;
    for (int i = 0; i < 40; ++i) s += __expf(x[b * 40 + i] - m);
    mx = m; ls = logf(s);
  }
  __syncthreads();
  for (int i = tid; i < 40; i += blockDim.x) out[b * 40 + i] = x[b * 40 + i] - mx - ls;
}

// ---------------- host side ---------------------------------------------------------
// Assumed input order: d_in[0] = xyz, then params leaves in alphabetical tree order:
// att1(fc1 w1,w2, fc2 w1,w2), att3(...), head(bn1_b,bn1_g,bn2_b,bn2_g,fc1_b,fc1_w,
// fc2_b,fc2_w,fc3_b,fc3_w), pscn(f,gu,gv,uv each w,b,g,be), sa1_b0 (3x w,b,g,be),
// sa1_b1 (3x), sa2 (3x).
enum {
  IN_XYZ = 0,
  IN_A1F1W1, IN_A1F1W2, IN_A1F2W1, IN_A1F2W2,
  IN_A3F1W1, IN_A3F1W2, IN_A3F2W1, IN_A3F2W2,
  IN_HD_BN1B, IN_HD_BN1G, IN_HD_BN2B, IN_HD_BN2G,
  IN_HD_FC1B, IN_HD_FC1W, IN_HD_FC2B, IN_HD_FC2W, IN_HD_FC3B, IN_HD_FC3W,
  IN_PF_W, IN_PF_B, IN_PF_G, IN_PF_BE,
  IN_PGU_W, IN_PGU_B, IN_PGU_G, IN_PGU_BE,
  IN_PGV_W, IN_PGV_B, IN_PGV_G, IN_PGV_BE,
  IN_PUV_W, IN_PUV_B, IN_PUV_G, IN_PUV_BE,
  IN_SA1B0 = 35,   // 3 layers x (w,b,g,be)
  IN_SA1B1 = 47,
  IN_SA2   = 59
};

static LayerH mkLH(const _Float16* wh, void* const* in, int base, int cinPad, int cout){
  LayerH L;
  L.w  = wh;
  L.b  = (const float*)in[base + 1];
  L.g  = (const float*)in[base + 2];
  L.be = (const float*)in[base + 3];
  L.cinPad = cinPad; L.cout = cout;
  return L;
}

extern "C" void kernel_launch(void* const* d_in, const int* in_sizes, int n_in,
                              void* d_out, int out_size, void* d_ws, size_t ws_size,
                              hipStream_t stream) {
  (void)in_sizes; (void)n_in; (void)out_size; (void)ws_size;
  const float* xyz = (const float*)d_in[IN_XYZ];
  float* ws = (float*)d_ws;

  // workspace bump allocation (floats)
  size_t cur = 0;
  auto alloc  = [&](size_t n){ float* p = ws + cur; cur += n; return p; };
  auto allocH = [&](size_t n){ return (_Float16*)alloc((n + 1) / 2); };
  float* coords = alloc(8 * 2048 * 3);
  float* norm   = alloc(8 * 2048 * 3);
  float* dist   = alloc(8 * 2048);
  int*   fidx   = (int*)alloc(8 * 256);
  float* newxyz = alloc(8 * 256 * 3);
  float* X0b0   = alloc((size_t)8 * 256 * 16 * 6);
  float* X0b1   = alloc((size_t)8 * 256 * 128 * 6);
  float* vbuf   = alloc((size_t)8 * 256 * 192);        // l1 (SA branch concat)
  int*   ides   = (int*)alloc(8 * 64);
  float* su     = alloc(8 * 64 * 3);
  float* sv     = alloc((size_t)8 * 64 * 192);
  float* Pu     = alloc(8 * 256 * 6);
  float* Qu     = alloc(8 * 64 * 6);
  float* Pv     = alloc((size_t)8 * 256 * 384);
  float* Qv     = alloc((size_t)8 * 64 * 384);
  float* hmax   = alloc((size_t)8 * 256 * 192);
  float* hf     = alloc((size_t)8 * 256 * 192);
  float* l1     = alloc((size_t)8 * 256 * 195);
  float* amean  = alloc(8 * 195);
  float* amax   = alloc(8 * 195);
  float* chat   = alloc(8 * 195);
  float* spat   = alloc(8 * 256);
  float* h198   = alloc((size_t)8 * 256 * 198);
  float* t1     = alloc((size_t)8 * 256 * 256);
  float* t2     = alloc((size_t)8 * 256 * 512);
  float* t3     = alloc((size_t)8 * 256 * 1024);
  float* l3     = alloc(8 * 1024);
  float* hd1    = alloc(8 * 512);
  float* hd2    = alloc(8 * 256);
  float* logits = alloc(8 * 40);
  // padded-f16 weight copies
  _Float16* b0w1h = allocH(32 * 32);
  _Float16* b0w2h = allocH(32 * 32);
  _Float16* b0w3h = allocH(64 * 32);
  _Float16* b1w1h = allocH(64 * 32);
  _Float16* b1w2h = allocH(96 * 64);
  _Float16* b1w3h = allocH(128 * 96);
  _Float16* gvwh  = allocH((size_t)384 * 384);
  _Float16* uvwh  = allocH((size_t)192 * 416);
  _Float16* fwh   = allocH((size_t)192 * 192);
  _Float16* s2w1h = allocH((size_t)256 * 224);
  _Float16* s2w2h = allocH((size_t)512 * 256);
  _Float16* s2w3h = allocH((size_t)1024 * 512);

  float* out_logits = (float*)d_out;
  float* out_l3     = (float*)d_out + 8 * 40;

  // weight pre-conversion to padded f16
  auto cvt = [&](int idx, _Float16* dst, int cout, int cin, int cinPad){
    int n = cout * cinPad;
    w2h_kernel<<<(n + 255) / 256, 256, 0, stream>>>((const float*)d_in[idx], dst, cout, cin, cinPad);
  };
  cvt(IN_SA1B0 + 0, b0w1h, 32, 6, 32);   cvt(IN_SA1B0 + 4, b0w2h, 32, 32, 32);
  cvt(IN_SA1B0 + 8, b0w3h, 64, 32, 32);
  cvt(IN_SA1B1 + 0, b1w1h, 64, 6, 32);   cvt(IN_SA1B1 + 4, b1w2h, 96, 64, 64);
  cvt(IN_SA1B1 + 8, b1w3h, 128, 96, 96);
  cvt(IN_PGV_W, gvwh, 384, 384, 384);
  cvt(IN_PUV_W, uvwh, 192, 390, 416);
  cvt(IN_PF_W,  fwh, 192, 192, 192);
  cvt(IN_SA2 + 0, s2w1h, 256, 198, 224);
  cvt(IN_SA2 + 4, s2w2h, 512, 256, 256);
  cvt(IN_SA2 + 8, s2w3h, 1024, 512, 512);

  // 0) transpose + split
  split_transpose_kernel<<<(8 * 2048 + 255) / 256, 256, 0, stream>>>(xyz, coords, norm);
  // 1) farthest point sampling -> 256 centroids / batch
  fps_kernel<<<8, 256, 0, stream>>>(coords, dist, fidx, newxyz);
  // 2) ball query + gather [norm ; dx] for both radii
  ballq_kernel<<<(8 * 256 + 255) / 256, 256, 0, stream>>>(coords, norm, newxyz, X0b0, 16, 0.1f * 0.1f);
  ballq_kernel<<<(8 * 256 + 255) / 256, 256, 0, stream>>>(coords, norm, newxyz, X0b1, 128, 0.4f * 0.4f);
  // 3) fused SA-branch MLP chains (WMMA) + max over neighborhood
  branch_mlp_kernel<<<8 * 256, 256, 0, stream>>>(X0b0, 16,
      mkLH(b0w1h, d_in, IN_SA1B0 + 0, 32, 32), mkLH(b0w2h, d_in, IN_SA1B0 + 4, 32, 32),
      mkLH(b0w3h, d_in, IN_SA1B0 + 8, 32, 64), vbuf, 192, 0);
  branch_mlp_kernel<<<8 * 256, 256, 0, stream>>>(X0b1, 128,
      mkLH(b1w1h, d_in, IN_SA1B1 + 0, 32, 64), mkLH(b1w2h, d_in, IN_SA1B1 + 4, 64, 96),
      mkLH(b1w3h, d_in, IN_SA1B1 + 8, 96, 128), vbuf, 192, 64);
  // 4) pscn: z-order sample, gathers, factorized gu/gv partial GEMMs
  zorder_kernel<<<8, 256, 0, stream>>>(newxyz, ides);
  gather_sample_kernel<<<(8 * 64 + 255) / 256, 256, 0, stream>>>(newxyz, vbuf, ides,
      (const float*)d_in[IN_PGU_W], su, sv, Qu);
  pu_kernel<<<(8 * 256 + 255) / 256, 256, 0, stream>>>(newxyz, (const float*)d_in[IN_PGU_W], Pu);
  {
    dim3 gp((2048 + 127) / 128, 384 / 64);
    gemm_f16w_kernel<<<gp, 256, 0, stream>>>(vbuf, 192, gvwh, 384, 0,
        nullptr, nullptr, nullptr, Pv, 384, 0, 2048, 192, 384, 0);
    dim3 gq((512 + 127) / 128, 384 / 64);
    gemm_f16w_kernel<<<gq, 256, 0, stream>>>(sv, 192, gvwh, 384, 192,
        nullptr, nullptr, nullptr, Qv, 384, 0, 512, 192, 384, 0);
  }
  // 5) fused uv GEMM + max over samples (dominant WMMA kernel)
  pscn_uv_kernel<<<8 * 256, 256, 0, stream>>>(Pu, Qu, Pv, Qv,
      (const float*)d_in[IN_PGU_B], (const float*)d_in[IN_PGU_G], (const float*)d_in[IN_PGU_BE],
      (const float*)d_in[IN_PGV_B], (const float*)d_in[IN_PGV_G], (const float*)d_in[IN_PGV_BE],
      mkLH(uvwh, d_in, IN_PUV_W, 416, 192), hmax);
  // 6) f layer + residual concat
  {
    dim3 gf((2048 + 127) / 128, 192 / 64);
    gemm_f16w_kernel<<<gf, 256, 0, stream>>>(hmax, 192, fwh, 192, 0,
        (const float*)d_in[IN_PF_B], (const float*)d_in[IN_PF_G], (const float*)d_in[IN_PF_BE],
        hf, 192, 0, 2048, 192, 192, 1);
  }
  l1_concat_kernel<<<(8 * 256 * 195 + 255) / 256, 256, 0, stream>>>(vbuf, hf, newxyz, l1);
  // 7) attention (channel + spatial), scale, concat with xyz
  meanmax_kernel<<<8, 256, 0, stream>>>(l1, amean, amax);
  att_kernel<<<8, 256, 0, stream>>>(amean, amax,
      (const float*)d_in[IN_A1F1W1], (const float*)d_in[IN_A1F1W2],
      (const float*)d_in[IN_A1F2W1], (const float*)d_in[IN_A1F2W2], chat, 195);
  att_kernel<<<8, 256, 0, stream>>>(amean, amax,
      (const float*)d_in[IN_A3F1W1], (const float*)d_in[IN_A3F1W2],
      (const float*)d_in[IN_A3F2W1], (const float*)d_in[IN_A3F2W2], spat, 256);
  scale_concat_kernel<<<(8 * 256 * 198 + 255) / 256, 256, 0, stream>>>(l1, chat, spat, newxyz, h198);
  // 8) sa2 MLP (WMMA) + global max pool
  {
    dim3 g1((2048 + 127) / 128, 256 / 64);
    gemm_f16w_kernel<<<g1, 256, 0, stream>>>(h198, 198, s2w1h, 224, 0,
        (const float*)d_in[IN_SA2 + 1], (const float*)d_in[IN_SA2 + 2], (const float*)d_in[IN_SA2 + 3],
        t1, 256, 0, 2048, 198, 256, 1);
    dim3 g2((2048 + 127) / 128, 512 / 64);
    gemm_f16w_kernel<<<g2, 256, 0, stream>>>(t1, 256, s2w2h, 256, 0,
        (const float*)d_in[IN_SA2 + 5], (const float*)d_in[IN_SA2 + 6], (const float*)d_in[IN_SA2 + 7],
        t2, 512, 0, 2048, 256, 512, 1);
    dim3 g3((2048 + 127) / 128, 1024 / 64);
    gemm_f16w_kernel<<<g3, 256, 0, stream>>>(t2, 512, s2w3h, 512, 0,
        (const float*)d_in[IN_SA2 + 9], (const float*)d_in[IN_SA2 + 10], (const float*)d_in[IN_SA2 + 11],
        t3, 1024, 0, 2048, 512, 1024, 1);
  }
  maxn_kernel<<<(8 * 1024 + 255) / 256, 256, 0, stream>>>(t3, l3, out_l3);
  // 9) head
  {
    dim3 gf1(8, (512 + 255) / 256);
    fc_kernel<<<gf1, 256, 0, stream>>>(l3, (const float*)d_in[IN_HD_FC1W], (const float*)d_in[IN_HD_FC1B],
        (const float*)d_in[IN_HD_BN1G], (const float*)d_in[IN_HD_BN1B], hd1, 1024, 512, 1);
    dim3 gf2(8, (256 + 255) / 256);
    fc_kernel<<<gf2, 256, 0, stream>>>(hd1, (const float*)d_in[IN_HD_FC2W], (const float*)d_in[IN_HD_FC2B],
        (const float*)d_in[IN_HD_BN2G], (const float*)d_in[IN_HD_BN2B], hd2, 512, 256, 1);
    dim3 gf3(8, 1);
    fc_kernel<<<gf3, 64, 0, stream>>>(hd2, (const float*)d_in[IN_HD_FC3W], (const float*)d_in[IN_HD_FC3B],
        nullptr, nullptr, logits, 256, 40, 0);
  }
  logsoftmax_kernel<<<8, 64, 0, stream>>>(logits, out_logits);
}
